// attention_61753039782220
// MI455X (gfx1250) — compile-verified, matrix operand order checked by reading, not run on hardware
//
#include <hip/hip_runtime.h>
#include <hip/hip_bf16.h>
#include <math.h>

// ---------------------------------------------------------------------------
// Bahdanau additive attention, fused for MI455X (gfx1250, wave32, WMMA).
// B=32, T=2048, H=U=1024 (hardcoded from the reference).
//
// Pipeline:
//   1) uaprep_kernel  : Ua fp32 -> bf16, pre-swizzled into the exact WMMA
//                       B-operand LDS image (2 MB, L2-resident).
//   2) decproj_kernel : dec_proj = dec_hidden @ Wa (tiny).
//   3) score_kernel   : fused enc_out@Ua GEMM (v_wmma_f32_16x16x32_bf16,
//                       fp32 accum) + tanh + Va dot -> score[B,T].
//                       B tiles double-buffered via
//                       global_load_async_to_lds_b128 (ASYNCcnt), K-loop
//                       interleave pinned with sched_group_barrier.
//   4) softmax_kernel : softmax over T.
//   5) context_kernel : context = alpha @ enc_out.
// ---------------------------------------------------------------------------

typedef __bf16 v16bf __attribute__((ext_vector_type(16)));
typedef float  v8f   __attribute__((ext_vector_type(8)));

#define BB 32
#define TT 2048
#define HH 1024
#define UU 1024

union Frag {
    v16bf v;
    uint4 q[2];
    unsigned int u[8];
};

__device__ inline unsigned short f32_to_bf16_rne(float f) {
    unsigned int b = __float_as_uint(f);
    unsigned int r = b + 0x7FFFu + ((b >> 16) & 1u);   // round to nearest even
    return (unsigned short)(r >> 16);
}

// WMMA 16-bit operand swizzle (ISA 7.12.2), 16x32 tile, element (row, kk):
//   lane = row + 16*((kk>>3)&1);  dword p = ((kk&7)>>1) + 4*(kk>>4);  half = kk&1
// Tiles are stored as [lane][8 dwords] so one lane's fragment = 2x ds_load_b128.
__device__ inline int swz_u32(int row, int kk_even) {  // u32 index, kk even
    int ln = row + (((kk_even >> 3) & 1) << 4);
    int p  = ((kk_even & 7) >> 1) + ((kk_even >> 4) << 2);
    return ln * 8 + p;
}

__device__ inline void frag_load(Frag& f, const unsigned int* base, int kb, int lane) {
    const uint4* p = (const uint4*)(base + kb * 256 + lane * 8);
    f.q[0] = p[0];
    f.q[1] = p[1];
}

// Issue one 32 KB pre-swizzled B tile as 16 async b128 DMAs per thread.
__device__ inline void async_copy_tile(unsigned int ldsDstAddr, const uint4* src,
                                       int tid) {
#pragma unroll
    for (int i = 0; i < 16; ++i) {
        int idx = i * 128 + tid;
        unsigned int la = ldsDstAddr + (unsigned int)(idx * 16);
        unsigned int va = (unsigned int)(idx * 16);
        asm volatile("global_load_async_to_lds_b128 %0, %1, %2"
                     :: "v"(la), "v"(va), "s"(src) : "memory");
    }
}

// ----------------- kernel 0: pre-swizzle Ua into WMMA B image --------------
// out layout (u32): [nt 0..63][kb 0..31][lane 0..31][p 0..7]
__global__ __launch_bounds__(256)
void uaprep_kernel(const float* __restrict__ Ua, unsigned int* __restrict__ out) {
    int flat = blockIdx.x * 256 + threadIdx.x;   // over (kpair, n): 512*1024
    int kp = flat >> 10;
    int n  = flat & 1023;
    int k  = kp * 2;
    float lo = Ua[(size_t)k * UU + n];
    float hi = Ua[(size_t)(k + 1) * UU + n];
    unsigned int pk = (unsigned int)f32_to_bf16_rne(lo)
                    | ((unsigned int)f32_to_bf16_rne(hi) << 16);
    int nt = n >> 4, n16 = n & 15;
    int kb = k >> 5, kk = k & 31;
    out[nt * 8192 + kb * 256 + swz_u32(n16, kk)] = pk;
}

// ------------------------- kernel 1: dec_proj ------------------------------
__global__ __launch_bounds__(256)
void decproj_kernel(const float* __restrict__ dec, const float* __restrict__ Wa,
                    float* __restrict__ dp) {
    int b = blockIdx.x;
    int u = blockIdx.y * 256 + threadIdx.x;
    const float* dh = dec + b * HH;
    float acc = 0.f;
    for (int h = 0; h < HH; ++h)
        acc = fmaf(dh[h], Wa[(size_t)h * UU + u], acc);   // dh[h] is wave-uniform
    dp[b * UU + u] = acc;
}

// --------------- kernel 2: fused enc_proj GEMM + tanh + Va dot -------------
// 128 threads = 4 waves. Each wave owns 32 M-rows (2 M-tiles) and shares one
// B fragment between 2 WMMAs (2 independent accumulator chains).
// LDS: A image 256 KB + double-buffered B tiles 64 KB = 320 KB (WGP max).
__global__ __launch_bounds__(128)
void score_kernel(const float* __restrict__ enc, const uint4* __restrict__ uaswz,
                  const float* __restrict__ dp, const float* __restrict__ Va,
                  float* __restrict__ score) {
    extern __shared__ unsigned int smem32[];
    unsigned int* ldsA  = smem32;             // [mtile 8][kb 32][lane 32][p 8] u32
    unsigned int* ldsB0 = smem32 + 65536;     // B tile buffer 0 (32 KB)
    unsigned int* ldsB1 = smem32 + 65536 + 8192;  // B tile buffer 1 (32 KB)

    const int tid  = threadIdx.x;
    const int lane = tid & 31;
    const int wave = tid >> 5;
    const int mBase = blockIdx.x * 128;       // 512 blocks x 128 rows
    const int bIdx  = mBase >> 11;            // 128 | 2048 -> one batch per block

    const unsigned int bAddr0 = (unsigned int)(uintptr_t)ldsB0;
    const unsigned int bAddr1 = (unsigned int)(uintptr_t)ldsB1;

    // Kick off DMA of B tile 0 immediately; it overlaps A staging below.
    async_copy_tile(bAddr0, uaswz, tid);

    // ---- Stage A once: 128 rows x 1024 K, fp32 -> bf16, packed u32 stores.
    for (int i = 0; i < 512; ++i) {
        int flat = i * 128 + tid;             // (m, kpair); coalesced along k
        int m  = flat >> 9;
        int kp = flat & 511;
        int k  = kp * 2;
        float2 v = ((const float2*)enc)[((size_t)(mBase + m) * HH + k) >> 1];
        unsigned int pk = (unsigned int)f32_to_bf16_rne(v.x)
                        | ((unsigned int)f32_to_bf16_rne(v.y) << 16);
        int mt = m >> 4, mi = m & 15;
        int kb = k >> 5, kk = k & 31;
        ldsA[mt * 8192 + kb * 256 + swz_u32(mi, kk)] = pk;
    }

    const unsigned int* aBase0 = ldsA + (2 * wave)     * 8192;
    const unsigned int* aBase1 = ldsA + (2 * wave + 1) * 8192;

    float sacc0[8], sacc1[8];
#pragma unroll
    for (int i = 0; i < 8; ++i) { sacc0[i] = 0.f; sacc1[i] = 0.f; }

    for (int nt = 0; nt < UU / 16; ++nt) {
        const int n0 = nt * 16;
        // All waves done with the buffer we are about to overwrite.
        __syncthreads();

        if (nt < UU / 16 - 1) {
            // DMA tile nt+1 into the other buffer; async loads complete in
            // order, so <=16 outstanding guarantees tile nt has landed.
            async_copy_tile((nt & 1) ? bAddr0 : bAddr1,
                            uaswz + (size_t)(nt + 1) * 2048, tid);
            asm volatile("s_wait_asynccnt 0x10" ::: "memory");
        } else {
            asm volatile("s_wait_asynccnt 0x0" ::: "memory");
        }
        __syncthreads();                      // tile nt visible to all waves

        const unsigned int* bBase = (nt & 1) ? ldsB1 : ldsB0;

        v8f c0, c1;
#pragma unroll
        for (int i = 0; i < 8; ++i) { c0[i] = 0.f; c1[i] = 0.f; }

        // ---- Software-pipelined K loop: 6 ds_load_b128 -> 2 WMMAs per kb.
        Frag a0, a1, b0, a0n, a1n, b0n;
        frag_load(a0, aBase0, 0, lane);
        frag_load(a1, aBase1, 0, lane);
        frag_load(b0, bBase,  0, lane);
#pragma unroll
        for (int kb = 0; kb < HH / 32; ++kb) {
            if (kb < HH / 32 - 1) {
                frag_load(a0n, aBase0, kb + 1, lane);
                frag_load(a1n, aBase1, kb + 1, lane);
                frag_load(b0n, bBase,  kb + 1, lane);
            }
            c0 = __builtin_amdgcn_wmma_f32_16x16x32_bf16(
                     false, a0.v, false, b0.v, (short)0, c0, false, false);
            c1 = __builtin_amdgcn_wmma_f32_16x16x32_bf16(
                     false, a1.v, false, b0.v, (short)0, c1, false, false);
            a0 = a0n; a1 = a1n; b0 = b0n;
        }
        // Pin the interleave: 12 DS reads up front (kb0+kb1 operands), then a
        // steady 2xWMMA -> 6xDS-read pattern so every WMMA has the next
        // iteration's loads already in flight (DS latency hidden).
        __builtin_amdgcn_sched_group_barrier(0x100, 12, 0);   // DS read
#pragma unroll
        for (int it = 0; it < 30; ++it) {
            __builtin_amdgcn_sched_group_barrier(0x008, 2, 0); // WMMA
            __builtin_amdgcn_sched_group_barrier(0x100, 6, 0); // DS read
        }
        __builtin_amdgcn_sched_group_barrier(0x008, 2, 0);
        __builtin_amdgcn_sched_group_barrier(0x008, 2, 0);

        // c[i] @ lane l -> row (i + 8*(l>=16)), col n0 + (l&15)
        float dpl = dp[bIdx * UU + n0 + (lane & 15)];   // L2-resident
        float vav = Va[n0 + (lane & 15)];
#pragma unroll
        for (int i = 0; i < 8; ++i) {
            sacc0[i] += tanhf(c0[i] + dpl) * vav;
            sacc1[i] += tanhf(c1[i] + dpl) * vav;
        }
    }

    // ---- Reduce over the 16 columns held across each 16-lane half.
#pragma unroll
    for (int i = 0; i < 8; ++i) {
        float v0 = sacc0[i], v1 = sacc1[i];
        for (int off = 1; off < 16; off <<= 1) {
            v0 += __shfl_xor(v0, off, 32);
            v1 += __shfl_xor(v1, off, 32);
        }
        sacc0[i] = v0; sacc1[i] = v1;
    }
    int rowBase = mBase + wave * 32;
    if (lane == 0) {
#pragma unroll
        for (int i = 0; i < 8; ++i) {
            score[rowBase + i]      = sacc0[i];
            score[rowBase + 16 + i] = sacc1[i];
        }
    } else if (lane == 16) {
#pragma unroll
        for (int i = 0; i < 8; ++i) {
            score[rowBase + 8 + i]  = sacc0[i];
            score[rowBase + 24 + i] = sacc1[i];
        }
    }
}

// ------------------------- kernel 3: softmax over T ------------------------
__global__ __launch_bounds__(256)
void softmax_kernel(const float* __restrict__ score, float* __restrict__ alpha) {
    __shared__ float red[256];
    int b = blockIdx.x, tid = threadIdx.x;
    const float* s = score + b * TT;
    float m = -INFINITY;
    for (int t = tid; t < TT; t += 256) m = fmaxf(m, s[t]);
    red[tid] = m; __syncthreads();
    for (int o = 128; o > 0; o >>= 1) {
        if (tid < o) red[tid] = fmaxf(red[tid], red[tid + o]);
        __syncthreads();
    }
    m = red[0]; __syncthreads();
    float sum = 0.f;
    for (int t = tid; t < TT; t += 256) sum += __expf(s[t] - m);
    red[tid] = sum; __syncthreads();
    for (int o = 128; o > 0; o >>= 1) {
        if (tid < o) red[tid] += red[tid + o];
        __syncthreads();
    }
    float inv = 1.f / red[0];
    for (int t = tid; t < TT; t += 256) alpha[b * TT + t] = __expf(s[t] - m) * inv;
}

// ------------------- kernel 4: context = alpha @ enc_out -------------------
__global__ __launch_bounds__(256)
void context_kernel(const float* __restrict__ alpha, const float* __restrict__ enc,
                    float* __restrict__ out) {
    int b = blockIdx.x;
    int h = blockIdx.y * 256 + threadIdx.x;
    const float* al = alpha + b * TT;
    const float* e  = enc + (size_t)b * TT * HH + h;
    float acc = 0.f;
    for (int t = 0; t < TT; ++t)
        acc = fmaf(al[t], e[(size_t)t * HH], acc);       // al[t] wave-uniform
    out[b * HH + h] = acc;
}

// ---------------------------------------------------------------------------
extern "C" void kernel_launch(void* const* d_in, const int* in_sizes, int n_in,
                              void* d_out, int out_size, void* d_ws, size_t ws_size,
                              hipStream_t stream) {
    (void)in_sizes; (void)n_in; (void)out_size; (void)ws_size;
    const float* enc = (const float*)d_in[0];   // [B,T,H]
    const float* dec = (const float*)d_in[1];   // [B,H]
    const float* Wa  = (const float*)d_in[2];   // [H,U]
    const float* Ua  = (const float*)d_in[3];   // [H,U]
    const float* Va  = (const float*)d_in[4];   // [U]
    float* out = (float*)d_out;                 // [B,H]

    float* dp    = (float*)d_ws;                // [B,U]   128 KB
    float* score = dp + BB * UU;                // [B,T]   256 KB
    float* alpha = score + BB * TT;             // [B,T]   256 KB
    unsigned int* uaswz = (unsigned int*)(alpha + BB * TT);  // 2 MB bf16 image

    uaprep_kernel<<<(512 * 1024) / 256, 256, 0, stream>>>(Ua, uaswz);
    decproj_kernel<<<dim3(BB, UU / 256), 256, 0, stream>>>(dec, Wa, dp);

    size_t smem = (size_t)(65536 + 2 * 8192) * sizeof(unsigned int);  // 320 KB
    score_kernel<<<(BB * TT) / 128, 128, smem, stream>>>(
        enc, (const uint4*)uaswz, dp, Va, score);

    softmax_kernel<<<BB, 256, 0, stream>>>(score, alpha);
    context_kernel<<<dim3(BB, HH / 256), 256, 0, stream>>>(alpha, enc, out);
}
